// GNetVerOne_58471684768027
// MI455X (gfx1250) — matrix operand order repaired and checked by reading, not compiled, not run on hardware
//
#include <hip/hip_runtime.h>
#include <hip/hip_bf16.h>

typedef __bf16 bf16_t;
typedef __attribute__((ext_vector_type(16))) __bf16 v16bf;
typedef __attribute__((ext_vector_type(8)))  __bf16 v8bf;
typedef __attribute__((ext_vector_type(4)))  __bf16 v4bf;
typedef __attribute__((ext_vector_type(8)))  float  v8f;

#define B_DIM   8
#define L_DIM   2048
#define K_NN    16
#define D_ATTR  64
#define NG      32
#define F_OUT   128
#define RDIM    2080        // 2048 (i*64+j) + 32 (embsum * K1)
#define RS      2088        // padded LDS row stride (bf16 elems) for A-panel
#define ES      33          // padded emb stride (floats)
#define AS      72          // padded attr stride (bf16 elems)

// LDS carve (bytes) -- total 144640 < 160KB => 2 workgroups per 320KB WGP
#define OFF_GKC   0          // 7*32 f32              = 896
#define OFF_GKP   896        // 7*7*32 f32            = 6272
#define OFF_EMB   7168       // 256*33 f32            = 33792
#define OFF_ATT   40960      // 256*72 bf16           = 36864
#define OFF_OUT   77824      // 16*2088 bf16          = 66816
#define SMEM_BYTES 144640

// ---------------------------------------------------------------------------
// Kernel 0: build transposed bf16 weight panel Wt[f][r], r = i*64+j for r<2048,
// r-2048 = i for the op_kernel1 tail rows.
// ---------------------------------------------------------------------------
__global__ void __launch_bounds__(256) wt_build_kernel(
    const float* __restrict__ k12, const float* __restrict__ k1,
    bf16_t* __restrict__ wt)
{
    int idx = blockIdx.x * 256 + threadIdx.x;
    if (idx >= F_OUT * RDIM) return;
    int f = idx / RDIM;
    int r = idx - f * RDIM;
    float v = (r < 2048) ? k12[(size_t)r * F_OUT + f]
                         : k1[(size_t)(r - 2048) * F_OUT + f];
    wt[(size_t)f * RDIM + r] = (bf16_t)v;
}

// ---------------------------------------------------------------------------
// Kernel 1: top-16 nearest neighbors per point (per batch), matching the
// reference d2 = |a|^2 + |b|^2 - 2 a.b formula and stable index tie-break.
// ---------------------------------------------------------------------------
__global__ void __launch_bounds__(256) knn_kernel(
    const float* __restrict__ frame, int* __restrict__ nidx)
{
    __shared__ float sc[L_DIM * 3];
    __shared__ float ssq[L_DIM];
    int b    = blockIdx.x >> 3;        // 8 query tiles of 256 per batch
    int tile = blockIdx.x & 7;
    int t    = threadIdx.x;
    const float* fb = frame + (size_t)b * L_DIM * 12;
    for (int i = t; i < L_DIM; i += 256) {
        float x = fb[i * 12 + 0], y = fb[i * 12 + 1], z = fb[i * 12 + 2];
        sc[i * 3 + 0] = x; sc[i * 3 + 1] = y; sc[i * 3 + 2] = z;
        ssq[i] = x * x + y * y + z * z;
    }
    __syncthreads();

    int q = tile * 256 + t;
    float qx = sc[q * 3], qy = sc[q * 3 + 1], qz = sc[q * 3 + 2], qs = ssq[q];

    float bd[K_NN]; int bi[K_NN];
#pragma unroll
    for (int j = 0; j < K_NN; ++j) { bd[j] = 3.4e38f; bi[j] = 0; }

    for (int m = 0; m < L_DIM; ++m) {
        float dot = qx * sc[m * 3] + qy * sc[m * 3 + 1] + qz * sc[m * 3 + 2];
        float d2  = qs + ssq[m] - 2.0f * dot;
        if (d2 < bd[K_NN - 1]) {
            float d = d2; int id = m;
#pragma unroll
            for (int j = 0; j < K_NN; ++j) {       // stable bubble insertion
                bool sw = d < bd[j];
                float td = bd[j]; int ti = bi[j];
                bd[j] = sw ? d  : bd[j];
                bi[j] = sw ? id : bi[j];
                d  = sw ? td : d;
                id = sw ? ti : id;
            }
        }
    }
    int base = ((b * L_DIM + q) << 4);
#pragma unroll
    for (int j = 0; j < K_NN; ++j) nidx[base + j] = bi[j];
}

// ---------------------------------------------------------------------------
// Kernel 2: fused gather + coords + gaussian embedding + outer-product panel
// + WMMA GEMM (16 positions x 2080) @ Wt (2080 x 128) + bias.
// 1 block = 16 positions, 256 threads (8 waves), one N-tile per wave.
// ---------------------------------------------------------------------------
__global__ void __launch_bounds__(256) fused_kernel(
    const float* __restrict__ frame, const float* __restrict__ attrs,
    const int*  __restrict__ aaidx, const float* __restrict__ gkc,
    const float* __restrict__ gkp,  const int*  __restrict__ nidx,
    const bf16_t* __restrict__ wt,  const float* __restrict__ bias,
    float* __restrict__ out)
{
    extern __shared__ char smem[];
    float*  s_gkc = (float*)(smem + OFF_GKC);    // [7][32]
    float*  s_gkp = (float*)(smem + OFF_GKP);    // [7][7][32]
    float*  s_emb = (float*)(smem + OFF_EMB);    // [256][ES] f32
    bf16_t* s_att = (bf16_t*)(smem + OFF_ATT);   // [256][AS] bf16
    bf16_t* s_out = (bf16_t*)(smem + OFF_OUT);   // [16][RS] A-panel bf16

    int tid   = threadIdx.x;
    int b     = blockIdx.x >> 7;                 // 128 tiles per batch
    int tile  = blockIdx.x & 127;
    int l0    = tile * 16;
    int bBase = b * L_DIM;

    for (int i = tid; i < 7 * NG; i += 256)      s_gkc[i] = gkc[i];
    for (int i = tid; i < 7 * 7 * NG; i += 256)  s_gkp[i] = gkp[i];
    __syncthreads();

    // ---------------- Phase A: per-(p,k) coords + gaussian embedding --------
    {
        int s = tid;                             // slot 0..255 = p*16 + k
        int p = s >> 4, k = s & 15;
        int gl = bBase + l0 + p;
        int n  = nidx[(gl << 4) + k];            // neighbor within batch
        const float* fs = frame + (size_t)gl * 12;
        const float* fn = frame + (size_t)(bBase + n) * 12;

        float dx = fn[0] - fs[0], dy = fn[1] - fs[1], dz = fn[2] - fs[2];
        float zsx = fs[9], zsy = fs[10], zsz = fs[11];
        float znx = fn[9], zny = fn[10], znz = fn[11];

        float c0 = dx * fs[3] + dy * fs[4] + dz * fs[5];
        float c1 = dx * fs[6] + dy * fs[7] + dz * fs[8];
        float c2 = dx * zsx + dy * zsy + dz * zsz;
        float c3 = fminf(fabsf((float)aaidx[bBase + n] - (float)aaidx[gl]), 8.0f);
        float c4 = znx * zsx + zny * zsy + znz * zsz;
        float dist = sqrtf(dx * dx + dy * dy + dz * dz + 1e-6f);
        float inv  = 1.0f / dist;
        float dnx = dx * inv, dny = dy * inv, dnz = dz * inv;
        float c5 = dnx * zsx + dny * zsy + dnz * zsz;
        float c6 = znx * dnx + zny * dny + znz * dnz;
        float crd[7] = {c0, c1, c2, c3, c4, c5, c6};

        float* eout = s_emb + s * ES;
        for (int g = 0; g < NG; ++g) {
            float diff[7];
#pragma unroll
            for (int d = 0; d < 7; ++d) diff[d] = crd[d] - s_gkc[d * NG + g];
            float acc = 0.0f;
#pragma unroll
            for (int e = 0; e < 7; ++e) {
                float y = 0.0f;
#pragma unroll
                for (int d = 0; d < 7; ++d) y += diff[d] * s_gkp[(d * 7 + e) * NG + g];
                acc += y * y;
            }
            eout[g] = __expf(-0.5f * acc);
        }

        const float4* arow = (const float4*)(attrs + (size_t)(bBase + n) * D_ATTR);
        bf16_t* adst = s_att + s * AS;
#pragma unroll
        for (int j = 0; j < 16; ++j) {
            float4 a = arow[j];
            v4bf w;
            w[0] = (bf16_t)a.x; w[1] = (bf16_t)a.y;
            w[2] = (bf16_t)a.z; w[3] = (bf16_t)a.w;
            *(v4bf*)(adst + j * 4) = w;
        }
    }
    __syncthreads();

    // ---------------- Phase B: build bf16 A-panel [16][2080] ----------------
    // outer[p][i*64+j] = sum_k emb[p,k,i] * attr[p,k,j]
    for (int gi = 0; gi < 32; ++gi) {
        int g   = tid + gi * 256;        // 0..8191 : 16p * 32i * 16 j-groups
        int p   = g >> 9;
        int rem = g & 511;
        int i   = rem >> 4;
        int j0  = (rem & 15) << 2;
        const float*  eb = s_emb + (p * 16) * ES + i;
        const bf16_t* at = s_att + (p * 16) * AS + j0;
        float a0 = 0.f, a1 = 0.f, a2 = 0.f, a3 = 0.f;
#pragma unroll
        for (int k = 0; k < K_NN; ++k) {
            float e = eb[k * ES];
            v4bf  a = *(const v4bf*)(at + k * AS);
            a0 += e * (float)a[0]; a1 += e * (float)a[1];
            a2 += e * (float)a[2]; a3 += e * (float)a[3];
        }
        bf16_t* dst = s_out + p * RS + i * 64 + j0;
        dst[0] = (bf16_t)a0; dst[1] = (bf16_t)a1;
        dst[2] = (bf16_t)a2; dst[3] = (bf16_t)a3;
    }
    // embsum tail columns 2048..2079 (drives the op_kernel1 term)
    for (int gi = 0; gi < 2; ++gi) {
        int g = tid + gi * 256;          // 0..511 = 16p * 32i
        int p = g >> 5, i = g & 31;
        const float* eb = s_emb + (p * 16) * ES + i;
        float ssum = 0.f;
#pragma unroll
        for (int k = 0; k < K_NN; ++k) ssum += eb[k * ES];
        s_out[p * RS + 2048 + i] = (bf16_t)ssum;
    }
    __syncthreads();

    // ---------------- Phase C: WMMA GEMM 16x128 = A(16x2080) @ Wt ----------
    int lane = tid & 31;
    int wave = tid >> 5;                  // 8 waves -> 8 x 16-col output tiles
    int mr = lane & 15;                   // A row / B column within tile
    int g8 = (lane < 16) ? 0 : 8;         // A-frag K-group offset
    int kb = (lane < 16) ? 0 : 8;         // B-frag K-group offset
    int f  = wave * 16 + mr;

    const bf16_t* arow = s_out + mr * RS + g8;
    const bf16_t* brow = wt + (size_t)f * RDIM + kb;

    v8f acc = {};
    for (int c = 0; c < RDIM / 32; ++c) {        // 65 K-chunks of 32
        union { v16bf v; struct { v8bf lo; v8bf hi; } s; } ua;
        ua.s.lo = *(const v8bf*)(arow + c * 32);
        ua.s.hi = *(const v8bf*)(arow + c * 32 + 16);
        union { v16bf v; struct { v8bf lo; v8bf hi; } s; } ub;
        ub.s.lo = *(const v8bf*)(brow + c * 32);
        ub.s.hi = *(const v8bf*)(brow + c * 32 + 16);
        v16bf bfv = ub.v;
        acc = __builtin_amdgcn_wmma_f32_16x16x32_bf16(false, ua.v, false, bfv,
                                                      (short)0, acc, false, false);
    }

    int mrow = (lane < 16) ? 0 : 8;       // C/D layout: rows r / r+8
    float bv = bias[f];
#pragma unroll
    for (int r = 0; r < 8; ++r) {
        size_t row = (size_t)(bBase + l0 + mrow + r);
        out[row * F_OUT + f] = acc[r] + bv;
    }
}

// ---------------------------------------------------------------------------
extern "C" void kernel_launch(void* const* d_in, const int* in_sizes, int n_in,
                              void* d_out, int out_size, void* d_ws, size_t ws_size,
                              hipStream_t stream) {
    const float* attrs = (const float*)d_in[0];   // aa_attributes (8,2048,64)
    const float* frame = (const float*)d_in[1];   // frame (8,2048,4,3)
    const int*   aaidx = (const int*)d_in[2];     // aa_indices (8,2048)
    // d_in[3] labels: unused
    const float* gkc  = (const float*)d_in[4];    // gk_centers (7,32)
    const float* gkp  = (const float*)d_in[5];    // gk_sqrt_prec (7,7,32)
    const float* k12  = (const float*)d_in[6];    // op_kernel12 (32,64,128)
    const float* k1   = (const float*)d_in[7];    // op_kernel1 (32,128)
    const float* bias = (const float*)d_in[8];    // op_bias (128,)
    float* out = (float*)d_out;

    int*    nidx = (int*)d_ws;                                    // 1 MiB
    bf16_t* wt   = (bf16_t*)((char*)d_ws + (size_t)B_DIM * L_DIM * K_NN * 4);

    wt_build_kernel<<<(F_OUT * RDIM + 255) / 256, 256, 0, stream>>>(k12, k1, wt);
    knn_kernel<<<B_DIM * (L_DIM / 256), 256, 0, stream>>>(frame, nidx);

    (void)hipFuncSetAttribute(reinterpret_cast<const void*>(fused_kernel),
                              hipFuncAttributeMaxDynamicSharedMemorySize,
                              SMEM_BYTES);
    fused_kernel<<<B_DIM * (L_DIM / 16), 256, SMEM_BYTES, stream>>>(
        frame, attrs, aaidx, gkc, gkp, nidx, wt, bias, out);
}
